// grud_model_5738076308227
// MI455X (gfx1250) — matrix operand order repaired and checked by reading, not compiled, not run on hardware
//
#include <hip/hip_runtime.h>
#include <cstdint>

typedef __attribute__((ext_vector_type(16))) _Float16 v16h;
typedef __attribute__((ext_vector_type(8)))  float    v8f;

constexpr int kB = 64, kC = 128, kH = 256, kO = 128, kT = 512;
constexpr int kTB = kT * kB;

struct alignas(16) U128 { unsigned int a, b, c, d; };
union V16U { v16h h; U128 q[2]; };
union V8FU { v8f v; float f[8]; };

// ---- WMMA tile loaders (CDNA5 16x16x32 f16 layouts, ISA 7.12.2) ------------

// A matrix 16x32 (MxK), row-major source with row stride lda (halves).
// lane L<16: M=L, K pairs {0..7} and {16..23}; lane L>=16: M=L-16, K {8..15},{24..31}
__device__ __forceinline__ v16h load_a_f16(const _Float16* A, int lda, int lane,
                                           int row0, int k0) {
  const int r  = row0 + (lane & 15);
  const int kb = k0 + ((lane & 16) ? 8 : 0);
  const _Float16* p = A + (size_t)r * lda + kb;
  V16U u;
  u.q[0] = *reinterpret_cast<const U128*>(p);       // K = kb .. kb+7
  u.q[1] = *reinterpret_cast<const U128*>(p + 16);  // K = kb+16 .. kb+23
  return u.h;
}

// B matrix 32x16 (KxN) from weight stored row-major [N][K] with row stride ldw.
// lane L<16: N=L, K {0..7},{16..23}; lane L>=16: N=L-16, K {8..15},{24..31}
__device__ __forceinline__ v16h load_b_f16(const _Float16* W, int ldw, int lane,
                                           int n0, int k0) {
  const int n  = n0 + (lane & 15);
  const int kb = k0 + ((lane & 16) ? 8 : 0);
  const _Float16* p = W + (size_t)n * ldw + kb;
  V16U u;
  u.q[0] = *reinterpret_cast<const U128*>(p);       // K = kb .. kb+7
  u.q[1] = *reinterpret_cast<const U128*>(p + 16);  // K = kb+16 .. kb+23
  return u.h;
}

__device__ __forceinline__ float sigmoidf_(float x) {
  return 1.0f / (1.0f + __expf(-x));
}

__device__ __forceinline__ v8f wmma_f16(v16h a, v16h b, v8f c) {
  return __builtin_amdgcn_wmma_f32_16x16x32_f16(false, a, false, b,
                                                (short)0, c, false, false);
}

// ---- small helper kernels ---------------------------------------------------

__global__ void __launch_bounds__(256)
f32_to_f16_kernel(const float* __restrict__ s, _Float16* __restrict__ d, int n) {
  int i = blockIdx.x * blockDim.x + threadIdx.x;
  if (i < n) d[i] = (_Float16)s[i];
}

// inp [B,3,C,T] f32  ->  time-major [T,B,C]: Xf (f32), M16/D16 (f16)
__global__ void __launch_bounds__(256)
reorder_inputs(const float* __restrict__ inp, float* __restrict__ Xf,
               _Float16* __restrict__ M16, _Float16* __restrict__ D16) {
  int i = blockIdx.x * blockDim.x + threadIdx.x;     // t fastest
  if (i >= kTB * kC) return;
  const int t = i % kT;
  const int c = (i / kT) % kC;
  const int b = i / (kT * kC);
  const size_t ib = ((size_t)(b * 3 + 0) * kC + c) * kT + t;
  const size_t im = ((size_t)(b * 3 + 1) * kC + c) * kT + t;
  const size_t id = ((size_t)(b * 3 + 2) * kC + c) * kT + t;
  const size_t o  = ((size_t)t * kB + b) * kC + c;
  Xf[o]  = inp[ib];
  M16[o] = (_Float16)inp[im];
  D16[o] = (_Float16)inp[id];
}

// store one 16x16 f32 tile with gamma activation: exp(-relu(v + bias))
__device__ __forceinline__ void store_gamma(float* __restrict__ out, int N,
                                            const float* __restrict__ bias,
                                            int row0, int col0, int lane,
                                            const V8FU& acc) {
  const int col = col0 + (lane & 15);
  const int rb  = (lane & 16) ? 8 : 0;
  const float bv = bias[col];
  #pragma unroll
  for (int i = 0; i < 8; ++i) {
    const float v = acc.f[i] + bv;
    out[(size_t)(row0 + rb + i) * N + col] = __expf(-fmaxf(v, 0.0f));
  }
}

// out[r, n] = exp(-relu( A @ W^T + bias )), A: [kTB, K]; 2x2 register blocking
__global__ void __launch_bounds__(256)
gemm_gamma(const _Float16* __restrict__ A, const _Float16* __restrict__ W,
           const float* __restrict__ bias, float* __restrict__ out,
           int N, int nt2 /*col tile-pairs*/, int K, int ldw) {
  const int wave = threadIdx.x >> 5;
  const int lane = threadIdx.x & 31;
  const int tile = blockIdx.x * 8 + wave;
  const int row0 = (tile / nt2) * 32;
  const int col0 = (tile % nt2) * 32;
  V8FU a00 = {}, a01 = {}, a10 = {}, a11 = {};
  for (int k0 = 0; k0 < K; k0 += 32) {
    v16h va0 = load_a_f16(A, K, lane, row0, k0);
    v16h va1 = load_a_f16(A, K, lane, row0 + 16, k0);
    v16h vb0 = load_b_f16(W, ldw, lane, col0, k0);
    v16h vb1 = load_b_f16(W, ldw, lane, col0 + 16, k0);
    a00.v = wmma_f16(va0, vb0, a00.v);
    a01.v = wmma_f16(va0, vb1, a01.v);
    a10.v = wmma_f16(va1, vb0, a10.v);
    a11.v = wmma_f16(va1, vb1, a11.v);
  }
  store_gamma(out, N, bias, row0,      col0,      lane, a00);
  store_gamma(out, N, bias, row0,      col0 + 16, lane, a01);
  store_gamma(out, N, bias, row0 + 16, col0,      lane, a10);
  store_gamma(out, N, bias, row0 + 16, col0 + 16, lane, a11);
}

// sequential-in-t last-observation scan + imputation; parallel over (b,c)
__global__ void __launch_bounds__(256)
scan_impute(const float* __restrict__ Xf, const _Float16* __restrict__ M16,
            const float* __restrict__ GX, const float* __restrict__ xmean_p,
            _Float16* __restrict__ XI) {
  const int bc = blockIdx.x * blockDim.x + threadIdx.x;   // [0, kB*kC)
  const float xmean = xmean_p[0];
  float x_last = 0.0f;
  for (int t = 0; t < kT; ++t) {
    const size_t idx = (size_t)t * kB * kC + bc;
    const float m = (float)M16[idx];
    const float x = Xf[idx];
    if (m > 0.0f) x_last = x;
    const float gx = GX[idx];
    const float xi = m * x + (1.0f - m) * (gx * x_last + (1.0f - gx) * xmean);
    XI[idx] = (_Float16)xi;
  }
}

__device__ __forceinline__ void store_pre(float* __restrict__ out,
                                          const float* __restrict__ bias,
                                          int row0, int col0, int lane,
                                          const V8FU& acc) {
  const int col = col0 + (lane & 15);
  const int rb  = (lane & 16) ? 8 : 0;
  const float bv = bias ? bias[col] : 0.0f;
  #pragma unroll
  for (int i = 0; i < 8; ++i)
    out[(size_t)(row0 + rb + i) * kH + col] = acc.f[i] + bv;
}

// pre_g = XI @ Wx_g^T + M @ Wm_g^T (+ bias_g); gate g = blockIdx.y; 2x2 blocked
__global__ void __launch_bounds__(256)
gemm_pre(const _Float16* __restrict__ XI, const _Float16* __restrict__ M,
         const _Float16* __restrict__ WxZ, const _Float16* __restrict__ WmZ,
         const _Float16* __restrict__ WxR, const _Float16* __restrict__ WmR,
         const _Float16* __restrict__ WxH, const _Float16* __restrict__ WmH,
         const float* __restrict__ bZ, const float* __restrict__ bH,
         float* __restrict__ preZ, float* __restrict__ preR,
         float* __restrict__ preH) {
  const int gate = blockIdx.y;
  const _Float16* Wx = (gate == 0) ? WxZ : (gate == 1) ? WxR : WxH;
  const _Float16* Wm = (gate == 0) ? WmZ : (gate == 1) ? WmR : WmH;
  const float* bias  = (gate == 0) ? bZ : (gate == 1) ? nullptr : bH;
  float* out         = (gate == 0) ? preZ : (gate == 1) ? preR : preH;

  const int wave = threadIdx.x >> 5;
  const int lane = threadIdx.x & 31;
  const int tile = blockIdx.x * 8 + wave;
  const int row0 = (tile / 8) * 32;   // kH/32 = 8 col tile-pairs
  const int col0 = (tile % 8) * 32;
  V8FU a00 = {}, a01 = {}, a10 = {}, a11 = {};
  #pragma unroll
  for (int k0 = 0; k0 < kC; k0 += 32) {
    v16h va0 = load_a_f16(XI, kC, lane, row0, k0);
    v16h va1 = load_a_f16(XI, kC, lane, row0 + 16, k0);
    v16h vb0 = load_b_f16(Wx, kC, lane, col0, k0);
    v16h vb1 = load_b_f16(Wx, kC, lane, col0 + 16, k0);
    a00.v = wmma_f16(va0, vb0, a00.v);
    a01.v = wmma_f16(va0, vb1, a01.v);
    a10.v = wmma_f16(va1, vb0, a10.v);
    a11.v = wmma_f16(va1, vb1, a11.v);
  }
  #pragma unroll
  for (int k0 = 0; k0 < kC; k0 += 32) {
    v16h va0 = load_a_f16(M, kC, lane, row0, k0);
    v16h va1 = load_a_f16(M, kC, lane, row0 + 16, k0);
    v16h vb0 = load_b_f16(Wm, kC, lane, col0, k0);
    v16h vb1 = load_b_f16(Wm, kC, lane, col0 + 16, k0);
    a00.v = wmma_f16(va0, vb0, a00.v);
    a01.v = wmma_f16(va0, vb1, a01.v);
    a10.v = wmma_f16(va1, vb0, a10.v);
    a11.v = wmma_f16(va1, vb1, a11.v);
  }
  store_pre(out, bias, row0,      col0,      lane, a00);
  store_pre(out, bias, row0,      col0 + 16, lane, a01);
  store_pre(out, bias, row0 + 16, col0,      lane, a10);
  store_pre(out, bias, row0 + 16, col0 + 16, lane, a11);
}

// ---- persistent recurrent kernel: one workgroup, h lives in LDS -------------
__global__ void __launch_bounds__(512)
grud_recurrent(const _Float16* __restrict__ wHZ, const _Float16* __restrict__ wHR,
               const _Float16* __restrict__ wHH, const _Float16* __restrict__ wHY,
               const float* __restrict__ bHY, const float* __restrict__ GH,
               const float* __restrict__ preZ, const float* __restrict__ preR,
               const float* __restrict__ preH,
               float* __restrict__ ys, float* __restrict__ hs) {
  extern __shared__ char smem[];
  float*    h32  = reinterpret_cast<float*>(smem);                 // 64 KB
  _Float16* h16  = reinterpret_cast<_Float16*>(smem + 65536);      // 32 KB
  _Float16* rh16 = reinterpret_cast<_Float16*>(smem + 98304);      // 32 KB
  _Float16* z16  = reinterpret_cast<_Float16*>(smem + 131072);     // 32 KB

  const int tid  = threadIdx.x;
  const int wave = tid >> 5;
  const int lane = tid & 31;
  const int rb   = (lane & 16) ? 8 : 0;
  const int cl   = lane & 15;

  for (int i = tid; i < kB * kH; i += 512) { h32[i] = 0.0f; h16[i] = (_Float16)0.0f; }
  __syncthreads();

  for (int t = 0; t < kT; ++t) {
    const size_t base = (size_t)t * kB * kH;

    // h <- gamma_h * h, keep f16 mirror fresh
    for (int i = tid; i < kB * kH; i += 512) {
      const float hv = h32[i] * GH[base + i];
      h32[i] = hv;
      h16[i] = (_Float16)hv;
    }
    if (t + 1 < kT) {  // global_prefetch_b8 next step's streams (64 KB each)
      const size_t nb = base + (size_t)kB * kH;
      __builtin_prefetch(preZ + nb + tid * 32, 0, 1);
      __builtin_prefetch(preR + nb + tid * 32, 0, 1);
      __builtin_prefetch(preH + nb + tid * 32, 0, 1);
      __builtin_prefetch(GH + nb + tid * 32, 0, 1);
    }
    __syncthreads();

    // Phase 1: z = sigmoid(preZ + h@Whz^T), r = sigmoid(preR + h@Whr^T)
    for (int tt = wave; tt < 128; tt += 16) {
      const bool isZ = tt < 64;
      const int t64  = tt & 63;
      const int row0 = (t64 & 3) * 16;
      const int col0 = (t64 >> 2) * 16;
      const float* pre   = (isZ ? preZ : preR) + base;
      const _Float16* W  = isZ ? wHZ : wHR;
      const int col = col0 + cl;
      V8FU acc;
      #pragma unroll
      for (int i = 0; i < 8; ++i)
        acc.f[i] = pre[(size_t)(row0 + rb + i) * kH + col];
      #pragma unroll
      for (int k0 = 0; k0 < kH; k0 += 32) {
        v16h a = load_a_f16(h16, kH, lane, row0, k0);
        v16h b = load_b_f16(W, kH, lane, col0, k0);
        acc.v = wmma_f16(a, b, acc.v);
      }
      #pragma unroll
      for (int i = 0; i < 8; ++i) {
        const int r = row0 + rb + i;
        const float s = sigmoidf_(acc.f[i]);
        if (isZ) z16[r * kH + col] = (_Float16)s;
        else     rh16[r * kH + col] = (_Float16)(s * h32[r * kH + col]);
      }
    }
    __syncthreads();

    // Phase 2: h_tilde = tanh(preH + (r*h)@Whh^T); h = (1-z)h + z*h_tilde
    for (int tt = wave; tt < 64; tt += 16) {
      const int row0 = (tt & 3) * 16;
      const int col0 = (tt >> 2) * 16;
      const float* pre = preH + base;
      const int col = col0 + cl;
      V8FU acc;
      #pragma unroll
      for (int i = 0; i < 8; ++i)
        acc.f[i] = pre[(size_t)(row0 + rb + i) * kH + col];
      #pragma unroll
      for (int k0 = 0; k0 < kH; k0 += 32) {
        v16h a = load_a_f16(rh16, kH, lane, row0, k0);
        v16h b = load_b_f16(wHH, kH, lane, col0, k0);
        acc.v = wmma_f16(a, b, acc.v);
      }
      #pragma unroll
      for (int i = 0; i < 8; ++i) {
        const int r = row0 + rb + i;
        const int idx = r * kH + col;
        const float ht = tanhf(acc.f[i]);
        const float z  = (float)z16[idx];
        const float hn = (1.0f - z) * h32[idx] + z * ht;
        h32[idx] = hn;
        h16[idx] = (_Float16)hn;
        hs[((size_t)r * kT + t) * kH + col] = hn;   // hs[B,T,H]
      }
    }
    __syncthreads();

    // Phase 3: y = sigmoid(h@Why^T + b_hy)
    for (int tt = wave; tt < 32; tt += 16) {
      const int row0 = (tt & 3) * 16;
      const int col0 = (tt >> 2) * 16;
      const int col = col0 + cl;
      V8FU acc;
      #pragma unroll
      for (int i = 0; i < 8; ++i) acc.f[i] = bHY[col];
      #pragma unroll
      for (int k0 = 0; k0 < kH; k0 += 32) {
        v16h a = load_a_f16(h16, kH, lane, row0, k0);
        v16h b = load_b_f16(wHY, kH, lane, col0, k0);
        acc.v = wmma_f16(a, b, acc.v);
      }
      #pragma unroll
      for (int i = 0; i < 8; ++i) {
        const int r = row0 + rb + i;
        ys[((size_t)r * kT + t) * kO + col] = sigmoidf_(acc.f[i]);  // ys[B,T,O]
      }
    }
    __syncthreads();
  }
}

// ---- host launch ------------------------------------------------------------

extern "C" void kernel_launch(void* const* d_in, const int* in_sizes, int n_in,
                              void* d_out, int out_size, void* d_ws, size_t ws_size,
                              hipStream_t stream) {
  const float* inp    = (const float*)d_in[0];
  const float* W_dg_x = (const float*)d_in[1];
  const float* b_dg_x = (const float*)d_in[2];
  const float* W_dg_h = (const float*)d_in[3];
  const float* b_dg_h = (const float*)d_in[4];
  const float* W_xz   = (const float*)d_in[5];
  const float* W_hz   = (const float*)d_in[6];
  const float* W_mz   = (const float*)d_in[7];
  const float* b_mz   = (const float*)d_in[8];
  const float* W_xr   = (const float*)d_in[9];
  const float* W_hr   = (const float*)d_in[10];
  const float* W_mr   = (const float*)d_in[11];
  const float* W_xh   = (const float*)d_in[12];
  const float* W_hh   = (const float*)d_in[13];
  const float* W_mh   = (const float*)d_in[14];
  const float* b_mh   = (const float*)d_in[15];
  const float* W_hy   = (const float*)d_in[16];
  const float* b_hy   = (const float*)d_in[17];
  const float* x_mean = (const float*)d_in[18];

  char* ws = (char*)d_ws;
  size_t off = 0;
  auto take = [&](size_t bytes) -> char* {
    char* p = ws + off;
    off = (off + bytes + 255) & ~(size_t)255;
    return p;
  };

  // f16 weights
  _Float16* wDGX = (_Float16*)take((size_t)kC * kC * 2);
  _Float16* wDGH = (_Float16*)take((size_t)kH * kC * 2);
  _Float16* wXZ  = (_Float16*)take((size_t)kH * kC * 2);
  _Float16* wMZ  = (_Float16*)take((size_t)kH * kC * 2);
  _Float16* wXR  = (_Float16*)take((size_t)kH * kC * 2);
  _Float16* wMR  = (_Float16*)take((size_t)kH * kC * 2);
  _Float16* wXH  = (_Float16*)take((size_t)kH * kC * 2);
  _Float16* wMH  = (_Float16*)take((size_t)kH * kC * 2);
  _Float16* wHZ  = (_Float16*)take((size_t)kH * kH * 2);
  _Float16* wHR  = (_Float16*)take((size_t)kH * kH * 2);
  _Float16* wHH  = (_Float16*)take((size_t)kH * kH * 2);
  _Float16* wHY  = (_Float16*)take((size_t)kO * kH * 2);
  // time-major activations
  float*    Xf   = (float*)take((size_t)kTB * kC * 4);
  _Float16* M16  = (_Float16*)take((size_t)kTB * kC * 2);
  _Float16* D16  = (_Float16*)take((size_t)kTB * kC * 2);
  float*    GX   = (float*)take((size_t)kTB * kC * 4);
  float*    GH   = (float*)take((size_t)kTB * kH * 4);
  _Float16* XI   = (_Float16*)take((size_t)kTB * kC * 2);
  float*    pZ   = (float*)take((size_t)kTB * kH * 4);
  float*    pR   = (float*)take((size_t)kTB * kH * 4);
  float*    pH   = (float*)take((size_t)kTB * kH * 4);

  auto conv = [&](const float* s, _Float16* d, int n) {
    f32_to_f16_kernel<<<(n + 255) / 256, 256, 0, stream>>>(s, d, n);
  };
  conv(W_dg_x, wDGX, kC * kC);
  conv(W_dg_h, wDGH, kH * kC);
  conv(W_xz, wXZ, kH * kC);  conv(W_mz, wMZ, kH * kC);
  conv(W_xr, wXR, kH * kC);  conv(W_mr, wMR, kH * kC);
  conv(W_xh, wXH, kH * kC);  conv(W_mh, wMH, kH * kC);
  conv(W_hz, wHZ, kH * kH);  conv(W_hr, wHR, kH * kH);
  conv(W_hh, wHH, kH * kH);  conv(W_hy, wHY, kO * kH);

  reorder_inputs<<<(kTB * kC + 255) / 256, 256, 0, stream>>>(inp, Xf, M16, D16);

  // gamma_x: [TB x C]; 2x2 blocking: (1024 row-pairs)*(4 col-pairs)/8 waves = 512
  gemm_gamma<<<512, 256, 0, stream>>>(D16, wDGX, b_dg_x, GX, kC, kC / 32, kC, kC);
  // gamma_h: [TB x H]; (1024)*(8)/8 = 1024 blocks
  gemm_gamma<<<1024, 256, 0, stream>>>(D16, wDGH, b_dg_h, GH, kH, kH / 32, kC, kC);

  scan_impute<<<(kB * kC) / 256, 256, 0, stream>>>(Xf, M16, GX, x_mean, XI);

  // (1024 row-pairs)*(8 col-pairs)/8 waves = 1024 blocks, 3 gates
  gemm_pre<<<dim3(1024, 3), 256, 0, stream>>>(XI, M16, wXZ, wMZ, wXR, wMR,
                                              wXH, wMH, b_mz, b_mh, pZ, pR, pH);

  float* ys = (float*)d_out;
  float* hs = ys + (size_t)kB * kT * kO;
  const size_t shmem = (size_t)kB * kH * 4 + 3 * (size_t)kB * kH * 2;  // 160 KB
  grud_recurrent<<<1, 512, shmem, stream>>>(wHZ, wHR, wHH, wHY, b_hy, GH,
                                            pZ, pR, pH, ys, hs);
}
